// CNN2D_simpleAttention_14267881358071
// MI455X (gfx1250) — matrix operand order repaired and checked by reading, not compiled, not run on hardware
//
#include <hip/hip_runtime.h>
#include <hip/hip_bf16.h>

#define B_     32
#define CIN    512
#define CENC   64
#define H_     56
#define W_     56
#define HW     (H_ * W_)      // 3136 = 196 * 16
#define K_     256
#define NTILES (HW / 16)      // 196 pixel tiles of 16
#define WG_PER_B 25           // 25 * 8 waves = 200 >= 196

typedef __attribute__((ext_vector_type(16))) __bf16 v16bf;
typedef __attribute__((ext_vector_type(8)))  float  v8f;

union Frag16 {
    uint4    q[2];
    unsigned u[8];
    v16bf    v;
};

// round-to-nearest-even fp32 -> bf16, pack two into a dword (a in low half)
static __device__ inline unsigned pack_bf16(float a, float b) {
    unsigned ua = __float_as_uint(a);
    unsigned ub = __float_as_uint(b);
    ua += 0x7FFFu + ((ua >> 16) & 1u);
    ub += 0x7FFFu + ((ub >> 16) & 1u);
    return (ua >> 16) | (ub & 0xFFFF0000u);
}

// ---------------------------------------------------------------------------
// Kernel 1: 1x1-conv GEMM via bf16 WMMA + fused bias + per-pixel norm.
// feats_out: [B, HW, 64] (pixel-major rows -> cheap gathers later)
// norms:     [B, HW]
// ---------------------------------------------------------------------------
__global__ __launch_bounds__(256) void conv_norm_kernel(
    const float* __restrict__ features,   // [B, CIN, HW]
    const float* __restrict__ conv_w,     // [CENC, CIN]
    const float* __restrict__ conv_b,     // [CENC]
    float* __restrict__ feats_out,
    float* __restrict__ norms)
{
    // bf16 weights resident in LDS: 64 x 512 x 2B = 64 KB (of 320 KB/WGP)
    __shared__ unsigned short Wlds[CENC * CIN];

    {   // cooperative fill: fp32 -> bf16
        unsigned* l32 = (unsigned*)Wlds;
        const float2* w2p = (const float2*)conv_w;
        for (int i = threadIdx.x; i < (CENC * CIN) / 2; i += 256) {
            float2 w2 = w2p[i];
            l32[i] = pack_bf16(w2.x, w2.y);
        }
    }
    __syncthreads();

    const int b    = blockIdx.x / WG_PER_B;
    const int wg   = blockIdx.x % WG_PER_B;
    const int wave = threadIdx.x >> 5;
    const int lane = threadIdx.x & 31;
    const int tIdx = wg * 8 + wave;
    if (tIdx >= NTILES) return;               // uniform per wave -> EXEC stays full

    const int p0  = tIdx * 16;                // 16 pixels for this wave
    const int nlo = lane & 15;
    const int h   = lane >> 4;                // lane half selects K-half

    v8f acc0 = {}, acc1 = {}, acc2 = {}, acc3 = {};

    // A rows: lane (mod 16) = pixel row; features are c-major (stride HW)
    const float* fbase = features + (size_t)b * CIN * HW + (p0 + nlo);

    for (int kk = 0; kk < CIN; kk += 32) {
        // ---- A fragment: 16x32 bf16 (contiguous K = 16h + s, matching B) ----
        Frag16 A;
#pragma unroll
        for (int j = 0; j < 8; ++j) {
            const int k0 = (j << 1) + h * 16;
            const float a0 = fbase[(size_t)(kk + k0)     * HW];
            const float a1 = fbase[(size_t)(kk + k0 + 1) * HW];
            A.u[j] = pack_bf16(a0, a1);
        }
        // ---- 4 N-tiles of 16 output channels each ----
#pragma unroll
        for (int t = 0; t < 4; ++t) {
            // B fragment: 32x16 bf16; lane = (Khalf,N); VGPR v holds K = 16h + 2v, 2v+1
            const unsigned short* wrow = &Wlds[(t * 16 + nlo) * CIN + kk + h * 16];
            Frag16 Bf;
            Bf.q[0] = *(const uint4*)(wrow);       // ds_load_b128 (K = 16h+0..7)
            Bf.q[1] = *(const uint4*)(wrow + 8);   // ds_load_b128 (K = 16h+8..15)
            v8f* accp = (t == 0) ? &acc0 : (t == 1) ? &acc1 : (t == 2) ? &acc2 : &acc3;
            *accp = __builtin_amdgcn_wmma_f32_16x16x32_bf16(
                        false, A.v, false, Bf.v, (short)0, *accp, false, false);
        }
    }

    // ---- epilogue: bias add, store feats [B,HW,64], accumulate row norms ----
    v8f nsum = {};
    float* frow = feats_out + (size_t)b * HW * CENC;
    v8f* accs[4] = { &acc0, &acc1, &acc2, &acc3 };
#pragma unroll
    for (int t = 0; t < 4; ++t) {
        const float bias = conv_b[t * 16 + nlo];
        v8f a = *accs[t];
#pragma unroll
        for (int r = 0; r < 8; ++r) {
            const float v   = a[r] + bias;           // D layout: row = p0 + r + 8h, col = t*16+nlo
            const int   row = p0 + r + h * 8;
            frow[(size_t)row * CENC + t * 16 + nlo] = v;
            nsum[r] += v * v;
        }
    }
    // xor-reduce over the 16 lanes of this half (bit4 separates row groups)
#pragma unroll
    for (int m = 1; m <= 8; m <<= 1) {
#pragma unroll
        for (int r = 0; r < 8; ++r) nsum[r] += __shfl_xor(nsum[r], m, 32);
    }
    if (nlo == 0) {
#pragma unroll
        for (int r = 0; r < 8; ++r)
            norms[(size_t)b * HW + p0 + h * 8 + r] = nsum[r];
    }
}

// ---------------------------------------------------------------------------
// Kernel 2: top-256 per batch. Bitonic sort (descending) of 4096 packed
// (norm_bits<<32 | pixel) keys in LDS; norms >= 0 so fp bit order is monotone.
// ---------------------------------------------------------------------------
__global__ __launch_bounds__(512) void topk_kernel(
    const float* __restrict__ norms, int* __restrict__ idx_out)
{
    __shared__ unsigned long long keys[4096];  // 32 KB
    const int b = blockIdx.x;

    for (int i = threadIdx.x; i < 4096; i += 512) {
        unsigned long long k = 0ull;
        if (i < HW) {
            const unsigned nb = __float_as_uint(norms[(size_t)b * HW + i]);
            k = ((unsigned long long)nb << 32) | (unsigned)i;
        }
        keys[i] = k;
    }
    __syncthreads();

    for (int size = 2; size <= 4096; size <<= 1) {
        for (int stride = size >> 1; stride > 0; stride >>= 1) {
            for (int t = threadIdx.x; t < 2048; t += 512) {
                const int i = 2 * stride * (t / stride) + (t % stride);
                const int j = i + stride;
                const bool up = (i & size) != 0;       // final pass: up=false -> descending
                const unsigned long long a = keys[i];
                const unsigned long long c = keys[j];
                const bool doswap = up ? (a > c) : (a < c);
                if (doswap) { keys[i] = c; keys[j] = a; }
            }
            __syncthreads();
        }
    }
    for (int t = threadIdx.x; t < K_; t += 512)
        idx_out[b * K_ + t] = (int)(unsigned)(keys[t] & 0xFFFFFFFFull);
}

// ---------------------------------------------------------------------------
// Kernel 3: zero the sparse attention map region of d_out
// ---------------------------------------------------------------------------
__global__ void zero_sw_kernel(float* __restrict__ sw, int n)
{
    const int i = blockIdx.x * 256 + threadIdx.x;
    if (i < n) sw[i] = 0.0f;
}

// ---------------------------------------------------------------------------
// Kernel 4: per (b,k) -> points row [x, y, 0, feat(64)] and scatter sw=1
// ---------------------------------------------------------------------------
__global__ __launch_bounds__(64) void points_kernel(
    const float* __restrict__ feats, const int* __restrict__ idx,
    float* __restrict__ sw, float* __restrict__ points)
{
    const int bk = blockIdx.x;           // b*K_ + k
    const int b  = bk >> 8;
    const int p  = idx[bk];
    const float* src = feats + ((size_t)b * HW + p) * CENC;
    float* dst = points + (size_t)bk * (3 + CENC);
    const int c = threadIdx.x;
    dst[3 + c] = src[c];
    if (c == 0) {
        dst[0] = (float)(p % W_);
        dst[1] = (float)(p / W_);
        dst[2] = 0.0f;
        sw[(size_t)b * HW + p] = 1.0f;
    }
}

// ---------------------------------------------------------------------------
// Kernel 5: x_out[b,c] = mean over K of gathered features
// ---------------------------------------------------------------------------
__global__ __launch_bounds__(64) void xout_kernel(
    const float* __restrict__ feats, const int* __restrict__ idx,
    float* __restrict__ xout)
{
    const int b = blockIdx.x;
    const int c = threadIdx.x;
    const int* ib = idx + b * K_;
    float s = 0.0f;
    for (int k = 0; k < K_; ++k)
        s += feats[((size_t)b * HW + ib[k]) * CENC + c];
    xout[b * CENC + c] = s * (1.0f / (float)K_);
}

// ---------------------------------------------------------------------------
extern "C" void kernel_launch(void* const* d_in, const int* in_sizes, int n_in,
                              void* d_out, int out_size, void* d_ws, size_t ws_size,
                              hipStream_t stream)
{
    const float* features = (const float*)d_in[0];   // [32,512,56,56]
    const float* conv_w   = (const float*)d_in[1];   // [64,512]
    const float* conv_b   = (const float*)d_in[2];   // [64]

    float* out    = (float*)d_out;
    float* xout   = out;                                   // 32*64        = 2048
    float* sw     = out + (size_t)B_ * CENC;               // 32*3136      = 100352
    float* points = sw  + (size_t)B_ * HW;                 // 32*256*67    = 548864

    float* feats = (float*)d_ws;                           // 32*3136*64 f32 (~25.7 MB)
    float* norms = feats + (size_t)B_ * HW * CENC;         // 32*3136 f32
    int*   idx   = (int*)(norms + (size_t)B_ * HW);        // 32*256 i32

    conv_norm_kernel<<<B_ * WG_PER_B, 256, 0, stream>>>(features, conv_w, conv_b,
                                                        feats, norms);
    topk_kernel<<<B_, 512, 0, stream>>>(norms, idx);
    zero_sw_kernel<<<(B_ * HW + 255) / 256, 256, 0, stream>>>(sw, B_ * HW);
    points_kernel<<<B_ * K_, 64, 0, stream>>>(feats, idx, sw, points);
    xout_kernel<<<B_, 64, 0, stream>>>(feats, idx, xout);
}
